// FoutLayer_54760833024352
// MI455X (gfx1250) — matrix operand order repaired and checked by reading, not compiled, not run on hardware
//
#include <hip/hip_runtime.h>
#include <hip/hip_bf16.h>

typedef _Float16 v16h __attribute__((ext_vector_type(16)));
typedef _Float16 v8h  __attribute__((ext_vector_type(8)));
typedef float    v8f  __attribute__((ext_vector_type(8)));
typedef float    v4f  __attribute__((ext_vector_type(4)));

#define IN_CH  512
#define OUT_CH 512
#define DEG    32
#define TM     128
#define TN     128
#define TK     32
#define LDSP   40   // f16 elems per LDS row: 32 + 8 pad = 80B (16B aligned)

// ---------------------------------------------------------------------------
// Kernel 0: transpose + fp32->f16 hi/lo split of both weight matrices.
// wc/wn are [K=512][C=512] row-major; outputs are [C][K] so GEMM B-tiles are
// K-contiguous per output column (matches WMMA B fragment layout).
// ---------------------------------------------------------------------------
__global__ __launch_bounds__(256) void prep_weights_kernel(
    const float* __restrict__ wc, const float* __restrict__ wn,
    _Float16* __restrict__ wct_h, _Float16* __restrict__ wct_l,
    _Float16* __restrict__ wnt_h, _Float16* __restrict__ wnt_l)
{
    int idx = blockIdx.x * blockDim.x + threadIdx.x;   // 0 .. 512*512-1
    int k = idx >> 9;
    int c = idx & 511;
    int tidx = c * IN_CH + k;

    float a = wc[idx];
    _Float16 h = (_Float16)a;
    _Float16 l = (_Float16)(a - (float)h);
    wct_h[tidx] = h; wct_l[tidx] = l;

    float b = wn[idx];
    h = (_Float16)b;
    l = (_Float16)(b - (float)h);
    wnt_h[tidx] = h; wnt_l[tidx] = l;
}

// ---------------------------------------------------------------------------
// WMMA fragment loads (layouts per CDNA5 ISA 7.12.2, wave32).
// A (16x32 f16): lane l: M = l&15; lanes<16 hold K={0..7,16..23}, lanes>=16
//                hold K={8..15,24..31}. Elements 0..7 = first K octet,
//                elements 8..15 = K+16 octet.
// B (32x16 f16): lane l: N = l&15; lanes<16 hold K={0..7,16..23}, lanes>=16
//                hold K={8..15,24..31} (same order as A).
// ---------------------------------------------------------------------------
__device__ inline v16h frag_a(const _Float16* base, int m, int kb) {
    const _Float16* p = base + m * LDSP + kb;
    v8h x0 = *(const v8h*)p;
    v8h x1 = *(const v8h*)(p + 16);
    v16h r;
#pragma unroll
    for (int i = 0; i < 8; ++i) { r[i] = x0[i]; r[i + 8] = x1[i]; }
    return r;
}
__device__ inline v16h frag_b(const _Float16* base, int n, int kb) {
    const _Float16* p = base + n * LDSP + kb;
    v8h x0 = *(const v8h*)p;
    v8h x1 = *(const v8h*)(p + 16);
    v16h r;
#pragma unroll
    for (int i = 0; i < 8; ++i) { r[i] = x0[i]; r[i + 8] = x1[i]; }
    return r;
}

// ---------------------------------------------------------------------------
// Kernel 1: fused dual GEMM via split-f16 WMMA (fp32-accurate:
// x@w ~= xh@wh + xh@wl + xl@wh, accumulated in f32).
// Grid: (8 col-tiles over [wc|wn], ceil(N/128) row-tiles). 256 thr = 8 waves.
// Each wave computes a 32x64 sub-tile = 2x4 WMMA 16x16 tiles.
// Col tiles 0-3 -> alpha (d_out), 4-7 -> beta (workspace).
// ---------------------------------------------------------------------------
__global__ __launch_bounds__(256) void gemm_split_kernel(
    const float* __restrict__ x,
    const _Float16* __restrict__ wct_h, const _Float16* __restrict__ wct_l,
    const _Float16* __restrict__ wnt_h, const _Float16* __restrict__ wnt_l,
    float* __restrict__ alpha_out,
    float* __restrict__ beta_out,
    int nrows)
{
    __shared__ _Float16 Ah[TM * LDSP];
    __shared__ _Float16 Al[TM * LDSP];
    __shared__ _Float16 Bh[TN * LDSP];
    __shared__ _Float16 Bl[TN * LDSP];

    const int t     = threadIdx.x;
    const int lane  = t & 31;
    const int wid   = t >> 5;         // 0..7
    const int waveM = wid & 3;        // 4 M groups of 32 rows
    const int waveN = wid >> 2;       // 2 N groups of 64 cols

    const int nt      = blockIdx.x;   // 0..7
    const int mBase   = blockIdx.y * TM;
    const int colBase = (nt & 3) * TN;
    const _Float16* bth = (nt < 4) ? wct_h : wnt_h;
    const _Float16* btl = (nt < 4) ? wct_l : wnt_l;
    float* dest = (nt < 4) ? alpha_out : beta_out;

    v8f acc[2][4];
    const v8f vzero = {0.f, 0.f, 0.f, 0.f, 0.f, 0.f, 0.f, 0.f};
#pragma unroll
    for (int mi = 0; mi < 2; ++mi)
#pragma unroll
        for (int ni = 0; ni < 4; ++ni)
            acc[mi][ni] = vzero;

    // A-tile fill mapping: thread -> (row = t>>1, 16-float segment = t&1)
    const int ar = t >> 1;
    const int as = (t & 1) * 16;
    int agrow = mBase + ar;
    if (agrow > nrows - 1) agrow = nrows - 1;   // clamp; stores are guarded
    const float* agp = x + (size_t)agrow * IN_CH + as;

    // B-tile fill mapping: two passes, thread -> (row = p*64 + t>>2, 8 f16)
    const int br = t >> 2;
    const int bj = (t & 3) * 8;

    const int kbA   = (lane < 16) ? 0 : 8;
    const int kbB   = (lane < 16) ? 0 : 8;
    const int mfrag = waveM * 32 + (lane & 15);
    const int nfrag = waveN * 64 + (lane & 15);

    for (int kt = 0; kt < IN_CH; kt += TK) {
        // ---- stage A (f32 -> hi/lo f16) ----
        {
            v4f f0 = *(const v4f*)(agp + kt);
            v4f f1 = *(const v4f*)(agp + kt + 4);
            v4f f2 = *(const v4f*)(agp + kt + 8);
            v4f f3 = *(const v4f*)(agp + kt + 12);
            v8h h0, h1, l0, l1;
#pragma unroll
            for (int i = 0; i < 4; ++i) {
                float v;
                _Float16 hh;
                v = f0[i]; hh = (_Float16)v; h0[i]     = hh; l0[i]     = (_Float16)(v - (float)hh);
                v = f1[i]; hh = (_Float16)v; h0[i + 4] = hh; l0[i + 4] = (_Float16)(v - (float)hh);
                v = f2[i]; hh = (_Float16)v; h1[i]     = hh; l1[i]     = (_Float16)(v - (float)hh);
                v = f3[i]; hh = (_Float16)v; h1[i + 4] = hh; l1[i + 4] = (_Float16)(v - (float)hh);
            }
            *(v8h*)(Ah + ar * LDSP + as)     = h0;
            *(v8h*)(Ah + ar * LDSP + as + 8) = h1;
            *(v8h*)(Al + ar * LDSP + as)     = l0;
            *(v8h*)(Al + ar * LDSP + as + 8) = l1;
        }
        // ---- stage B (already f16 hi/lo, K-contiguous rows) ----
#pragma unroll
        for (int p = 0; p < 2; ++p) {
            int row = p * 64 + br;
            const _Float16* gh = bth + (size_t)(colBase + row) * IN_CH + kt + bj;
            const _Float16* gl = btl + (size_t)(colBase + row) * IN_CH + kt + bj;
            v8h vh = *(const v8h*)gh;
            v8h vl = *(const v8h*)gl;
            *(v8h*)(Bh + row * LDSP + bj) = vh;
            *(v8h*)(Bl + row * LDSP + bj) = vl;
        }
        __syncthreads();

        // ---- WMMA compute ----
        v16h a_h[2], a_l[2];
#pragma unroll
        for (int mi = 0; mi < 2; ++mi) {
            a_h[mi] = frag_a(Ah, mfrag + mi * 16, kbA);
            a_l[mi] = frag_a(Al, mfrag + mi * 16, kbA);
        }
#pragma unroll
        for (int ni = 0; ni < 4; ++ni) {
            v16h b_h = frag_b(Bh, nfrag + ni * 16, kbB);
            v16h b_l = frag_b(Bl, nfrag + ni * 16, kbB);
#pragma unroll
            for (int mi = 0; mi < 2; ++mi) {
                acc[mi][ni] = __builtin_amdgcn_wmma_f32_16x16x32_f16(
                    false, a_h[mi], false, b_h, (short)0, acc[mi][ni], false, false);
                acc[mi][ni] = __builtin_amdgcn_wmma_f32_16x16x32_f16(
                    false, a_h[mi], false, b_l, (short)0, acc[mi][ni], false, false);
                acc[mi][ni] = __builtin_amdgcn_wmma_f32_16x16x32_f16(
                    false, a_l[mi], false, b_h, (short)0, acc[mi][ni], false, false);
            }
        }
        __syncthreads();
    }

    // ---- epilogue: C layout -> VGPR r holds M = r + 8*(lane>=16), N = lane&15
    const int rowHalf = (lane >> 4) * 8;
    const int coln    = colBase + waveN * 64 + (lane & 15);
#pragma unroll
    for (int mi = 0; mi < 2; ++mi) {
#pragma unroll
        for (int ni = 0; ni < 4; ++ni) {
#pragma unroll
            for (int r = 0; r < 8; ++r) {
                int row = mBase + waveM * 32 + mi * 16 + rowHalf + r;
                if (row < nrows)
                    dest[(size_t)row * OUT_CH + coln + ni * 16] = acc[mi][ni][r];
            }
        }
    }
}

// ---------------------------------------------------------------------------
// Kernel 2: gamma = mean over 32 neighbors of beta; out = alpha + gamma + bias.
// src = repeat(arange(N), 32) -> node n owns edges [32n, 32n+32).
// One block per node; 128 threads x float4 = 512 channels.
// beta (102 MB) lives in the 192 MB L2, so the 3.3 GB gather hits L2.
// ---------------------------------------------------------------------------
__global__ __launch_bounds__(128) void gather_kernel(
    const float* __restrict__ beta, const int* __restrict__ dstIdx,
    const float* __restrict__ bias, float* __restrict__ out)
{
    __shared__ int sd[DEG];
    const int n = blockIdx.x;
    const int t = threadIdx.x;
    if (t < DEG) sd[t] = dstIdx[n * DEG + t];
    __syncthreads();

    const int c4 = t * 4;
    v4f acc = {0.f, 0.f, 0.f, 0.f};
#pragma unroll
    for (int e = 0; e < DEG; ++e) {
        const v4f b = *(const v4f*)(beta + (size_t)sd[e] * OUT_CH + c4);
        acc += b;
    }
    const v4f al = *(const v4f*)(out  + (size_t)n * OUT_CH + c4);
    const v4f bi = *(const v4f*)(bias + c4);
    v4f r = al + bi + acc * (1.0f / (float)DEG);
    *(v4f*)(out + (size_t)n * OUT_CH + c4) = r;
}

// ---------------------------------------------------------------------------
extern "C" void kernel_launch(void* const* d_in, const int* in_sizes, int n_in,
                              void* d_out, int out_size, void* d_ws, size_t ws_size,
                              hipStream_t stream) {
    const float* x    = (const float*)d_in[0];
    const float* wc   = (const float*)d_in[1];
    const float* wn   = (const float*)d_in[2];
    const float* bias = (const float*)d_in[3];
    const int*   edge = (const int*)d_in[4];
    float* out = (float*)d_out;

    const int N = in_sizes[0] / IN_CH;
    const int E = in_sizes[4] / 2;
    const int* dstIdx = edge + E;   // edge_index row 0 = src, row 1 = dst

    // Workspace layout: beta[N][512] f32, then 4 transposed/split f16 weights.
    char* ws = (char*)d_ws;
    float* beta = (float*)ws;
    size_t off = (size_t)N * OUT_CH * sizeof(float);
    const size_t wbytes = (size_t)IN_CH * OUT_CH * sizeof(_Float16);
    _Float16* wct_h = (_Float16*)(ws + off); off += wbytes;
    _Float16* wct_l = (_Float16*)(ws + off); off += wbytes;
    _Float16* wnt_h = (_Float16*)(ws + off); off += wbytes;
    _Float16* wnt_l = (_Float16*)(ws + off); off += wbytes;

    prep_weights_kernel<<<(IN_CH * OUT_CH) / 256, 256, 0, stream>>>(
        wc, wn, wct_h, wct_l, wnt_h, wnt_l);

    dim3 grid(8, (N + TM - 1) / TM);
    gemm_split_kernel<<<grid, 256, 0, stream>>>(
        x, wct_h, wct_l, wnt_h, wnt_l, out, beta, N);

    gather_kernel<<<N, 128, 0, stream>>>(beta, dstIdx, bias, out);
}
